// MoELayer_62560493633925
// MI455X (gfx1250) — matrix operand order repaired and checked by reading, not compiled, not run on hardware
//
#include <hip/hip_runtime.h>

// ---------------------------------------------------------------------------
// MoE top-2 layer for gfx1250 (MI455X), wave32, v_wmma_f32_16x16x32_bf16.
// Weights converted fp32->bf16 AND transposed once (L2-resident copies), so
// FFN LDS staging is a straight b32 copy; fragments assemble as aligned
// ds_load_b32 pairs matching the documented A(16x32)/B(32x16) VGPR layouts.
// FFN block: 32 tokens x 1024 out, 512 threads = 16 waves (2 M-tiles x 8).
// ---------------------------------------------------------------------------

typedef __attribute__((ext_vector_type(16))) __bf16 v16bf;
typedef __attribute__((ext_vector_type(8)))  float  v8f;
typedef unsigned short us;
typedef unsigned int   u32;

#define N_TOK   4096   // SEQ * BATCH
#define DM      1024   // D_MODEL
#define DF      4096   // D_FF
#define NE      8      // experts
#define TILE_T  32     // tokens per ffn block
#define TILES_E 128    // N_TOK / TILE_T (worst-case tiles per expert)

// LDS partition (bytes) inside the ffn kernel (strides padded so that the
// 16 row-lanes of fragment loads land on distinct banks: 517/65/17/69 dwords)
#define LDS_X_OFF   0              // 32 x 1034 bf16  = 66176 B
#define LDS_W1_OFF  66176         // 128 x 130 bf16  = 33280 B ([f][k])
#define LDS_W2_OFF  99456         // 1024 x 34 bf16  = 69632 B ([d][k])
#define LDS_H_OFF   169088        // 32 x 138 bf16   =  8832 B
#define LDS_TOTAL   177920

__device__ __forceinline__ us f32_bf16(float f) {
  u32 u = __float_as_uint(f);
  u += 0x7fffu + ((u >> 16) & 1u);   // round-to-nearest-even
  return (us)(u >> 16);
}

union Frag { v16bf v; u32 u[8]; };

// A fragment (16x32 bf16) from row-major LDS [m][k], stride lda (even).
// lanes 0-15/16-31 hold M=lane&15; VGPR p holds K pair:
//   K = (p<4 ? 2p : 2p+8) + 8*(lane>>4)
__device__ __forceinline__ v16bf load_a_frag(const us* lds, int lda, int lane) {
  const int m = lane & 15, half = lane >> 4;
  Frag f;
#pragma unroll
  for (int p = 0; p < 8; ++p) {
    const int k = ((p < 4) ? (2 * p) : (2 * p + 8)) + half * 8;
    f.u[p] = *(const u32*)(lds + m * lda + k);
  }
  return f.v;
}

// B fragment (32x16 bf16) from K-transposed LDS [n][k], stride ldb (even).
// lanes 0-15/16-31 hold N=lane&15; VGPR p holds K pair:
//   K = (p<4 ? 2p : 2p+8) + 8*(lane>>4)
__device__ __forceinline__ v16bf load_b_frag(const us* ldsT, int ldb, int lane) {
  const int n = lane & 15, half = lane >> 4;
  Frag f;
#pragma unroll
  for (int p = 0; p < 8; ++p) {
    const int k = ((p < 4) ? (2 * p) : (2 * p + 8)) + half * 8;
    f.u[p] = *(const u32*)(ldsT + n * ldb + k);
  }
  return f.v;
}

// ---------------------------------------------------------------------------
// fp32 -> bf16 bulk convert (4 elems / thread) — used for x
// ---------------------------------------------------------------------------
__global__ __launch_bounds__(256) void cvt_bf16_kernel(
    const float* __restrict__ in, us* __restrict__ out, int n4) {
  int i = blockIdx.x * blockDim.x + threadIdx.x;
  if (i >= n4) return;
  const float4 v = ((const float4*)in)[i];
  u32 lo = (u32)f32_bf16(v.x) | ((u32)f32_bf16(v.y) << 16);
  u32 hi = (u32)f32_bf16(v.z) | ((u32)f32_bf16(v.w) << 16);
  ((u32*)out)[i * 2 + 0] = lo;
  ((u32*)out)[i * 2 + 1] = hi;
}

// ---------------------------------------------------------------------------
// fp32 [mat][R][C] -> bf16 [mat][C][R] transpose via 32x32 LDS tile.
// ---------------------------------------------------------------------------
__global__ __launch_bounds__(256) void cvt_t_kernel(
    const float* __restrict__ in, us* __restrict__ out, int R, int C) {
  __shared__ float tile[32][33];
  const int per_m = (R >> 5) * (C >> 5);
  const int m  = blockIdx.x / per_m;
  const int t  = blockIdx.x % per_m;
  const int tpr = C >> 5;
  const int tr = (t / tpr) << 5;
  const int tc = (t % tpr) << 5;
  const float* in_m = in + (size_t)m * R * C;
  us* out_m = out + (size_t)m * R * C;
  const int tid = threadIdx.x;
#pragma unroll
  for (int i = 0; i < 4; ++i) {
    int lin = tid + 256 * i;
    int r = lin >> 5, c = lin & 31;
    tile[r][c] = in_m[(size_t)(tr + r) * C + tc + c];
  }
  __syncthreads();
#pragma unroll
  for (int i = 0; i < 4; ++i) {
    int lin = tid + 256 * i;
    int c = lin >> 5, r = lin & 31;          // write rows of the transpose
    out_m[(size_t)(tc + c) * R + tr + r] = f32_bf16(tile[r][c]);
  }
}

// ---------------------------------------------------------------------------
// Gating: one wave32 per token. fp32 scores, top-2, softmax, append to lists.
// ---------------------------------------------------------------------------
__global__ __launch_bounds__(256) void gate_kernel(
    const float* __restrict__ x, const float* __restrict__ gw,
    const float* __restrict__ gb, int* __restrict__ cnt,
    int* __restrict__ list_tok, float* __restrict__ list_p) {
  const int wave = (blockIdx.x * blockDim.x + threadIdx.x) >> 5;
  const int lane = threadIdx.x & 31;
  if (wave >= N_TOK) return;

  float acc[NE];
#pragma unroll
  for (int e = 0; e < NE; ++e) acc[e] = 0.0f;

  const float* xr = x + (size_t)wave * DM;
  for (int d = lane; d < DM; d += 32) {
    const float xv = xr[d];
#pragma unroll
    for (int e = 0; e < NE; ++e) acc[e] += xv * gw[d * NE + e];
  }
#pragma unroll
  for (int e = 0; e < NE; ++e)
    for (int off = 16; off > 0; off >>= 1)
      acc[e] += __shfl_xor(acc[e], off, 32);

  if (lane == 0) {
    float s[NE];
#pragma unroll
    for (int e = 0; e < NE; ++e) s[e] = acc[e] + gb[e];
    int i0 = 0;
#pragma unroll
    for (int e = 1; e < NE; ++e) if (s[e] > s[i0]) i0 = e;  // first-max tie-break
    int i1 = -1;
#pragma unroll
    for (int e = 0; e < NE; ++e)
      if (e != i0 && (i1 < 0 || s[e] > s[i1])) i1 = e;
    const float v0 = s[i0], v1 = s[i1];
    const float p0 = 1.0f / (1.0f + expf(v1 - v0));   // softmax over {v0,v1}
    const float p1 = 1.0f - p0;

    int pos0 = atomicAdd(&cnt[i0], 1);
    list_tok[i0 * N_TOK + pos0] = wave * 2 + 0;
    list_p [i0 * N_TOK + pos0] = p0;
    int pos1 = atomicAdd(&cnt[i1], 1);
    list_tok[i1 * N_TOK + pos1] = wave * 2 + 1;
    list_p [i1 * N_TOK + pos1] = p1;
  }
}

// ---------------------------------------------------------------------------
// Fused expert FFN. Block = (expert, 32-token tile), 512 threads = 16 waves.
// Waves 0-7 own token rows 0-15 (mtile 0), waves 8-15 own rows 16-31.
// GEMM1: h[32x128] = relu(x_tile @ w1[e][:,fc:fc+128] + b1)  per f-chunk,
//        K staged 128 wide -> 4 WMMAs per barrier pair.
// GEMM2: y[32x1024] += h @ w2[e][fc:fc+128,:]  (8 v8f accumulators / wave).
// w1tg = bf16 w1^T [e][f][d];  w2tg = bf16 w2^T [e][d][f]  (K contiguous).
// ---------------------------------------------------------------------------
__global__ __launch_bounds__(512, 1) void ffn_kernel(
    const us* __restrict__ xb, const us* __restrict__ w1tg,
    const us* __restrict__ w2tg, const float* __restrict__ b1,
    const float* __restrict__ b2, const int* __restrict__ cnt,
    const int* __restrict__ list_tok, const float* __restrict__ list_p,
    float* __restrict__ ybuf) {
  const int e    = blockIdx.x >> 7;        // expert
  const int t    = blockIdx.x & 127;       // tile within expert
  const int nt   = cnt[e];
  const int base = t * TILE_T;
  if (base >= nt) return;                  // uniform exit: EXEC all-ones below

  extern __shared__ char smem[];
  us* lds_x   = (us*)(smem + LDS_X_OFF);   // [32][1034]
  us* lds_w1t = (us*)(smem + LDS_W1_OFF);  // [128][130]  rows f, cols k=d
  us* lds_w2t = (us*)(smem + LDS_W2_OFF);  // [1024][34]  rows d, cols k=f
  us* lds_h   = (us*)(smem + LDS_H_OFF);   // [32][138]

  const int tid   = threadIdx.x;
  const int lane  = tid & 31;
  const int w     = tid >> 5;              // wave id 0..15
  const int mtile = w >> 3;                // 0 or 1 (token half)
  const int wsub  = w & 7;                 // column-strip id 0..7

  // --- stage x tile (gather selected token rows, bf16) ---
  for (int r = w; r < TILE_T; r += 16) {
    int idx = base + r;
    int cl  = (idx < nt) ? idx : (nt - 1);             // replicate for padding
    int tok = list_tok[e * N_TOK + cl] >> 1;
    const u32* src = (const u32*)(xb + (size_t)tok * DM);
    u32* dst = (u32*)(lds_x + r * 1034);
    for (int c = lane; c < DM / 2; c += 32) dst[c] = src[c];
  }
  __syncthreads();

  v8f yacc[8];
#pragma unroll
  for (int j = 0; j < 8; ++j) yacc[j] = (v8f){0,0,0,0,0,0,0,0};

  for (int fc = 0; fc < DF; fc += 128) {
    // ================= GEMM1: h = relu(x @ w1 + b1) =================
    v8f hacc = (v8f){0,0,0,0,0,0,0,0};
    for (int kb = 0; kb < DM; kb += 128) {
      __syncthreads();                                  // lds_w1t reuse
      // prefetch next 128-wide k-block of w1^T (one 128B line per thread)
      if ((kb + 128 < DM) && (tid < 256))
        __builtin_prefetch(w1tg + ((size_t)(e * DF + fc + (tid >> 1))) * DM +
                               kb + 128 + (tid & 1) * 64, 0, 1);
      // straight copy: w1^T rows f=fc..fc+127, cols d=kb..kb+127 -> [f][k]
      {
        u32* dstw = (u32*)lds_w1t;
#pragma unroll 4
        for (int i = 0; i < 16; ++i) {
          int lin = tid + 512 * i;                      // 0..8191 (u32 units)
          int ff  = lin >> 6;                           // 0..127
          int kw  = lin & 63;                           // u32 col (2 bf16)
          const u32* src =
              (const u32*)(w1tg + ((size_t)(e * DF + fc + ff)) * DM + kb);
          dstw[ff * 65 + kw] = src[kw];
        }
      }
      __syncthreads();
#pragma unroll
      for (int kk = 0; kk < 4; ++kk) {
        v16bf a = load_a_frag(lds_x + mtile * 16 * 1034 + kb + kk * 32,
                              1034, lane);
        v16bf b = load_b_frag(lds_w1t + (wsub * 16) * 130 + kk * 32,
                              130, lane);
        hacc = __builtin_amdgcn_wmma_f32_16x16x32_bf16(
                   false, a, false, b, (short)0, hacc, false, false);
      }
    }
    // bias + relu + store h to LDS (C layout: n = lane&15, m = r + 8*(lane>>4))
    {
      const int n = lane & 15, half = lane >> 4;
      const float bias = b1[e * DF + fc + wsub * 16 + n];
      __syncthreads();                                  // lds_h reuse
#pragma unroll
      for (int r = 0; r < 8; ++r) {
        const int m = mtile * 16 + r + half * 8;
        float hv = hacc[r] + bias;
        hv = hv > 0.0f ? hv : 0.0f;
        lds_h[m * 138 + wsub * 16 + n] = f32_bf16(hv);
      }
      __syncthreads();
    }
    // ================= GEMM2: y += h @ w2 =================
    for (int ks = 0; ks < 4; ++ks) {
      __syncthreads();                                  // lds_w2t reuse
      // prefetch next f-block of w2^T (2 rows per thread)
      {
        int nf = fc + ((ks + 1) << 5);
        if (nf < DF) {
#pragma unroll
          for (int q = 0; q < 2; ++q)
            __builtin_prefetch(
                w2tg + ((size_t)(e * DM + tid * 2 + q)) * DF + nf, 0, 1);
        }
      }
      // straight copy: w2^T rows d=0..1023, cols f=fc+ks*32..+31 -> [d][k]
      {
        u32* dstw = (u32*)lds_w2t;
#pragma unroll 4
        for (int i = 0; i < 32; ++i) {
          int lin = tid + 512 * i;                      // 0..16383 (u32 units)
          int dd  = lin >> 4;                           // 0..1023
          int kw  = lin & 15;
          const u32* src =
              (const u32*)(w2tg + ((size_t)(e * DM + dd)) * DF + fc + (ks << 5));
          dstw[dd * 17 + kw] = src[kw];
        }
      }
      __syncthreads();
      v16bf a2 = load_a_frag(lds_h + mtile * 16 * 138 + ks * 32, 138, lane);
#pragma unroll
      for (int j = 0; j < 8; ++j) {
        v16bf bf = load_b_frag(lds_w2t + (wsub * 128 + j * 16) * 34, 34, lane);
        yacc[j] = __builtin_amdgcn_wmma_f32_16x16x32_bf16(
                      false, a2, false, bf, (short)0, yacc[j], false, false);
      }
    }
  }

  // --- weighted scatter: ybuf[(tok*2+slot)][d] = p * (y + b2[e][d]) ---
  {
    const int n = lane & 15, half = lane >> 4;
#pragma unroll
    for (int j = 0; j < 8; ++j) {
      const int d = wsub * 128 + j * 16 + n;
      const float bias2 = b2[e * DM + d];
#pragma unroll
      for (int r = 0; r < 8; ++r) {
        const int m   = mtile * 16 + r + half * 8;
        const int idx = base + m;
        if (idx < nt) {
          const int   packed = list_tok[e * N_TOK + idx];  // tok*2 + slot
          const float p      = list_p [e * N_TOK + idx];
          ybuf[(size_t)packed * DM + d] = p * (yacc[j][r] + bias2);
        }
      }
    }
  }
}

// ---------------------------------------------------------------------------
// out[n][d] = ybuf[n*2+0][d] + ybuf[n*2+1][d]
// ---------------------------------------------------------------------------
__global__ __launch_bounds__(256) void combine_kernel(
    const float* __restrict__ ybuf, float* __restrict__ out, int n) {
  int i = blockIdx.x * blockDim.x + threadIdx.x;
  if (i >= n) return;
  int tok = i >> 10, d = i & (DM - 1);
  out[i] = ybuf[(size_t)(tok * 2 + 0) * DM + d] +
           ybuf[(size_t)(tok * 2 + 1) * DM + d];
}

// ---------------------------------------------------------------------------
extern "C" void kernel_launch(void* const* d_in, const int* in_sizes, int n_in,
                              void* d_out, int out_size, void* d_ws, size_t ws_size,
                              hipStream_t stream) {
  (void)in_sizes; (void)n_in; (void)out_size; (void)ws_size;
  const float* x  = (const float*)d_in[0];
  const float* gw = (const float*)d_in[1];
  const float* gb = (const float*)d_in[2];
  const float* w1 = (const float*)d_in[3];
  const float* b1 = (const float*)d_in[4];
  const float* w2 = (const float*)d_in[5];
  const float* b2 = (const float*)d_in[6];
  float* out = (float*)d_out;

  // workspace layout (bytes)
  char* ws = (char*)d_ws;
  us*    xb   = (us*)   (ws + 0);           //   8,388,608  x bf16
  us*    w1tg = (us*)   (ws + 8388608);     //  67,108,864  w1^T bf16 [e][f][d]
  us*    w2tg = (us*)   (ws + 75497472);    //  67,108,864  w2^T bf16 [e][d][f]
  float* ybuf = (float*)(ws + 142606336);   //  33,554,432  per-(token,slot) y
  int*   ltok = (int*)  (ws + 176160768);   //     131,072  per-expert token lists
  float* lp   = (float*)(ws + 176291840);   //     131,072  per-expert probs
  int*   cnt  = (int*)  (ws + 176422912);   //          32  per-expert counts

  hipMemsetAsync(cnt, 0, NE * sizeof(int), stream);

  // x -> bf16 (row-major, A-operand layout needs no transpose)
  cvt_bf16_kernel<<<(N_TOK * DM / 4 + 255) / 256, 256, 0, stream>>>(
      x, xb, N_TOK * DM / 4);
  // w1 [e][D][F] -> bf16 w1^T [e][F][D];  w2 [e][F][D] -> bf16 w2^T [e][D][F]
  cvt_t_kernel<<<NE * (DM / 32) * (DF / 32), 256, 0, stream>>>(w1, w1tg, DM, DF);
  cvt_t_kernel<<<NE * (DF / 32) * (DM / 32), 256, 0, stream>>>(w2, w2tg, DF, DM);

  // routing (one wave per token, 8 waves per block)
  gate_kernel<<<N_TOK / 8, 256, 0, stream>>>(x, gw, gb, cnt, ltok, lp);

  // expert FFN, fixed worst-case grid with uniform early exit
  ffn_kernel<<<NE * TILES_E, 512, LDS_TOTAL, stream>>>(
      xb, w1tg, w2tg, b1, b2, cnt, ltok, lp, ybuf);

  // final top-2 combine
  combine_kernel<<<(N_TOK * DM + 255) / 256, 256, 0, stream>>>(
      ybuf, out, N_TOK * DM);
}
